// NoAudioQuantizer_11922829214093
// MI455X (gfx1250) — compile-verified, matrix operand order checked by reading, not run on hardware
//
#include <hip/hip_runtime.h>

// Problem dims (B=4, L=2048 -> BL=8192 rows)
#define BL    8192
#define DIMD  1280
#define CDIM  256
#define NCB   4096
#define NT    (NCB / 16)   // 256 codebook tiles

typedef __attribute__((ext_vector_type(16))) _Float16 v16h;
typedef __attribute__((ext_vector_type(8)))  float    v8f;
typedef __attribute__((ext_vector_type(8)))  _Float16 h8;

// D = A(16x32 f16) * B(32x16 f16) + C(f32) — probe-verified builtin
static __device__ __forceinline__ v8f wmma16(v16h a, v16h b, v8f c) {
  return __builtin_amdgcn_wmma_f32_16x16x32_f16(
      /*neg_a=*/false, a, /*neg_b=*/false, b,
      /*c_mod=*/(short)0, c, /*reuse_a=*/false, /*reuse_b=*/false);
}

// 16-bit fragment layout (ISA 7.12.2): element i of the 16-vector covers
// K = g*8 + i (i<8) and K = 16 + g*8 + (i-8) (i>=8)  where g = lane>>4.
// => each half is 8 contiguous K elements: two b128 loads per f16 fragment.
static __device__ __forceinline__ v16h frag_from_f16(const _Float16* __restrict__ rowp, int g) {
  const h8 lo = *(const h8*)(rowp + (g << 3));        // K = g*8 .. g*8+7
  const h8 hi = *(const h8*)(rowp + 16 + (g << 3));   // K = 16+g*8 ..
  v16h o;
#pragma unroll
  for (int i = 0; i < 8; ++i) { o[i] = lo[i]; o[8 + i] = hi[i]; }
  return o;
}

// fp32 source: 4x b128 loads + pk converts (used only where data is read once).
static __device__ __forceinline__ v16h frag_from_f32(const float* __restrict__ rowp, int g) {
  const float4 a0 = *(const float4*)(rowp + (g << 3));
  const float4 a1 = *(const float4*)(rowp + (g << 3) + 4);
  const float4 b0 = *(const float4*)(rowp + 16 + (g << 3));
  const float4 b1 = *(const float4*)(rowp + 16 + (g << 3) + 4);
  const float tmp[16] = {a0.x, a0.y, a0.z, a0.w, a1.x, a1.y, a1.z, a1.w,
                         b0.x, b0.y, b0.z, b0.w, b1.x, b1.y, b1.z, b1.w};
  v16h o;
#pragma unroll
  for (int i = 0; i < 16; ++i) o[i] = (_Float16)tmp[i];
  return o;
}

// CDNA5 async memory->LDS copy, 16B per lane (tracked with ASYNCcnt).
static __device__ __forceinline__ void async_copy_b128(unsigned lds_off,
                                                       const _Float16* gptr) {
  asm volatile("global_load_async_to_lds_b128 %0, %1, off"
               :: "v"(lds_off), "v"(gptr) : "memory");
}

// ---------------------------------------------------------------------------
// Kernel 0: elementwise fp32 -> f16 (weights; one-time, read-many payoff).
// 256 threads x 8 elements per block.
// ---------------------------------------------------------------------------
__global__ __launch_bounds__(256)
void k_cvt_f16(const float* __restrict__ src, _Float16* __restrict__ dst) {
  const size_t i = ((size_t)blockIdx.x * 256 + threadIdx.x) * 8;
  const float4 a = *(const float4*)(src + i);
  const float4 b = *(const float4*)(src + i + 4);
  h8 o;
  o[0] = (_Float16)a.x; o[1] = (_Float16)a.y; o[2] = (_Float16)a.z; o[3] = (_Float16)a.w;
  o[4] = (_Float16)b.x; o[5] = (_Float16)b.y; o[6] = (_Float16)b.z; o[7] = (_Float16)b.w;
  *(h8*)(dst + i) = o;
}

// ---------------------------------------------------------------------------
// Kernel 1: h = z @ W_in^T + b_in   (8192x1280 * 1280x256)
// One wave per 16x128 tile: A fragment (z, converted in-flight) reused across
// 8 f16-weight B fragments -> 8 WMMAs per A load.
// ---------------------------------------------------------------------------
__global__ __launch_bounds__(256)
void k_proj_in(const float* __restrict__ z, const _Float16* __restrict__ w16in,
               const float* __restrict__ b_in, float* __restrict__ hid,
               _Float16* __restrict__ h16) {
  const int wave = (blockIdx.x << 3) + (threadIdx.x >> 5);  // 1024 waves
  const int lane = threadIdx.x & 31;
  const int m0 = (wave >> 1) << 4;   // 512 m-tiles
  const int n0 = (wave & 1) << 7;    // 2 n-groups of 128
  const int col = lane & 15, g = lane >> 4;

  const float*    zrow = z     + (size_t)(m0 + col) * DIMD;
  const _Float16* wrow = w16in + (size_t)(n0 + col) * DIMD;
  v8f acc[8] = {};
  for (int k0 = 0; k0 < DIMD; k0 += 32) {
    const v16h a = frag_from_f32(zrow + k0, g);
#pragma unroll
    for (int tI = 0; tI < 8; ++tI)
      acc[tI] = wmma16(a, frag_from_f16(wrow + (size_t)(tI * 16) * DIMD + k0, g), acc[tI]);
  }
#pragma unroll
  for (int tI = 0; tI < 8; ++tI) {
    const int n = n0 + tI * 16 + col;
    const float bv = b_in[n];
#pragma unroll
    for (int r = 0; r < 8; ++r) {
      const int m = m0 + r + (g << 3);
      const float v = acc[tI][r] + bv;
      hid[(size_t)m * CDIM + n] = v;
      h16[(size_t)m * CDIM + n] = (_Float16)v;
    }
  }
}

// ---------------------------------------------------------------------------
// Kernel 2: per-codebook-row ||c||^2 and fp32->f16 codebook copy.
// ---------------------------------------------------------------------------
__global__ __launch_bounds__(256)
void k_csq(const float* __restrict__ cb, float* __restrict__ c_sq,
           _Float16* __restrict__ cb16) {
  const int row  = (blockIdx.x << 3) + (threadIdx.x >> 5);
  const int lane = threadIdx.x & 31;
  const float* p = cb + (size_t)row * CDIM + lane * 8;
  float s = 0.f;
  h8 pk;
#pragma unroll
  for (int i = 0; i < 8; ++i) { const float v = p[i]; s += v * v; pk[i] = (_Float16)v; }
  *(h8*)(cb16 + (size_t)row * CDIM + lane * 8) = pk;
#pragma unroll
  for (int off = 16; off >= 1; off >>= 1) s += __shfl_xor(s, off, 32);
  if (lane == 0) c_sq[row] = s;
}

// ---------------------------------------------------------------------------
// Kernel 3: indices = argmin_n ( c_sq[n] - 2 * (h . codebook[n]) ).
// Block = 8 waves x 16 rows = 128 rows; all waves sweep the SAME codebook
// tile, staged once per block into double-buffered LDS via
// global_load_async_to_lds_b128 (+ s_wait_asynccnt). A fragments (full K)
// stay in registers; B fragments come from LDS (padded stride, ds_load_b128).
// ---------------------------------------------------------------------------
__global__ __launch_bounds__(256)
void k_argmin(const _Float16* __restrict__ h16, const _Float16* __restrict__ cb16,
              const float* __restrict__ c_sq, int* __restrict__ idx_out) {
  // 264-halfword row stride = 132 dwords ≡ 4 (mod 64) -> conflict-free cols.
  __shared__ __align__(16) _Float16 tile[2][16][264];

  const int t = threadIdx.x;
  const int lane = t & 31;
  const int col = lane & 15, g = lane >> 4;
  const int m0 = blockIdx.x * 128 + (t >> 5) * 16;   // 64 blocks -> 512 m-tiles

  const int ldr = t >> 4;          // staging: row 0..15
  const int ldc = (t & 15) << 4;   // staging: 16-element (32B) chunk

  // A fragments: this wave's 16 h-rows, full K=256, kept in 64 VGPRs.
  v16h afr[8];
  const _Float16* hrow = h16 + (size_t)(m0 + col) * CDIM;
#pragma unroll
  for (int kt = 0; kt < 8; ++kt) afr[kt] = frag_from_f16(hrow + kt * 32, g);

  float best[8];
  int   bidx[8];
#pragma unroll
  for (int r = 0; r < 8; ++r) { best[r] = 3.4e38f; bidx[r] = 0; }

  // prologue: stage codebook tile 0 into buffer 0
  async_copy_b128((unsigned)(size_t)&tile[0][ldr][ldc],
                  cb16 + (size_t)ldr * CDIM + ldc);

  for (int nt = 0; nt < NT; ++nt) {
    const int cur = nt & 1;
    if (nt) __syncthreads();                 // everyone done with buf cur^1
    if (nt + 1 < NT) {
      async_copy_b128((unsigned)(size_t)&tile[cur ^ 1][ldr][ldc],
                      cb16 + (size_t)((nt + 1) * 16 + ldr) * CDIM + ldc);
      asm volatile("s_wait_asynccnt 0x1" ::: "memory");  // tile `cur` landed
    } else {
      asm volatile("s_wait_asynccnt 0x0" ::: "memory");
    }
    __syncthreads();                         // whole tile visible to all waves

    v8f acc = {};
#pragma unroll
    for (int kt = 0; kt < 8; ++kt)
      acc = wmma16(afr[kt], frag_from_f16(&tile[cur][col][kt * 32], g), acc);

    const int n = (nt << 4) + col;
    const float cq = c_sq[n];
#pragma unroll
    for (int r = 0; r < 8; ++r) {
      const float s = fmaf(-2.f, acc[r], cq);
      if (s < best[r]) { best[r] = s; bidx[r] = n; }  // strict < => first hit
    }
  }

  // fold the 16 lanes of each half onto their common rows
#pragma unroll
  for (int off = 8; off >= 1; off >>= 1) {
#pragma unroll
    for (int r = 0; r < 8; ++r) {
      const float ob = __shfl_xor(best[r], off, 32);
      const int   oi = __shfl_xor(bidx[r], off, 32);
      if (ob < best[r] || (ob == best[r] && oi < bidx[r])) { best[r] = ob; bidx[r] = oi; }
    }
  }
  if (col == 0) {
#pragma unroll
    for (int r = 0; r < 8; ++r) idx_out[m0 + r + (g << 3)] = bidx[r];
  }
}

// ---------------------------------------------------------------------------
// Kernel 4: out = codebook[idx] @ W_out^T + b_out, masked.
// One wave per 16x128 tile; gathered A fragments (K=256) loaded once and
// reused across 8 N-tiles (64 WMMAs per gather); f16 weights.
// ---------------------------------------------------------------------------
__global__ __launch_bounds__(256)
void k_proj_out(const int* __restrict__ idx, const _Float16* __restrict__ cb16,
                const _Float16* __restrict__ w16out, const float* __restrict__ b_out,
                const unsigned char* __restrict__ mask, float* __restrict__ out) {
  const int wave = (blockIdx.x << 3) + (threadIdx.x >> 5);  // 5120 waves
  const int lane = threadIdx.x & 31;
  const int m0 = (wave / 10) << 4;   // 512 m-tiles
  const int n0 = (wave % 10) << 7;   // 10 n-groups of 128
  const int col = lane & 15, g = lane >> 4;

  const int ci = idx[m0 + col];
  const _Float16* arow = cb16 + (size_t)ci * CDIM;
  v16h afr[8];
#pragma unroll
  for (int kt = 0; kt < 8; ++kt) afr[kt] = frag_from_f16(arow + kt * 32, g);

  unsigned char mk[8];
#pragma unroll
  for (int r = 0; r < 8; ++r) mk[r] = mask[m0 + r + (g << 3)];

#pragma unroll
  for (int tI = 0; tI < 8; ++tI) {
    const int d = n0 + tI * 16 + col;
    const _Float16* wrow = w16out + (size_t)d * CDIM;
    v8f acc = {};
#pragma unroll
    for (int kt = 0; kt < 8; ++kt)
      acc = wmma16(afr[kt], frag_from_f16(wrow + kt * 32, g), acc);
    const float bv = b_out[d];
#pragma unroll
    for (int r = 0; r < 8; ++r) {
      const int m = m0 + r + (g << 3);
      out[(size_t)m * DIMD + d] = mk[r] ? (acc[r] + bv) : 0.f;
    }
  }
}

// ---------------------------------------------------------------------------
extern "C" void kernel_launch(void* const* d_in, const int* in_sizes, int n_in,
                              void* d_out, int out_size, void* d_ws, size_t ws_size,
                              hipStream_t stream) {
  (void)in_sizes; (void)n_in; (void)out_size; (void)ws_size;
  const float*         z     = (const float*)d_in[0];
  const unsigned char* mask  = (const unsigned char*)d_in[1];   // bool (B,L)
  const float*         W_in  = (const float*)d_in[2];
  const float*         b_in  = (const float*)d_in[3];
  const float*         W_out = (const float*)d_in[4];
  const float*         b_out = (const float*)d_in[5];
  const float*         cb    = (const float*)d_in[6];

  // d_out = [ out (BL*D f32) | indices (BL i32) | hiddens (BL*C f32) ]
  float* out = (float*)d_out;
  int*   idx = (int*)((float*)d_out + (size_t)BL * DIMD);
  float* hid = (float*)d_out + (size_t)BL * DIMD + BL;

  // workspace: h16 (4 MB) | cb16 (2 MB) | c_sq (16 KB) | w16in | w16out
  char* ws = (char*)d_ws;
  _Float16* h16    = (_Float16*)ws;                                   ws += (size_t)BL * CDIM * 2;
  _Float16* cb16   = (_Float16*)ws;                                   ws += (size_t)NCB * CDIM * 2;
  float*    csq    = (float*)ws;                                      ws += (size_t)NCB * 4;
  _Float16* w16in  = (_Float16*)ws;                                   ws += (size_t)CDIM * DIMD * 2;
  _Float16* w16out = (_Float16*)ws;

  const int wblk = (CDIM * DIMD) / 2048;  // 160 blocks (256 thr x 8 elems)
  k_cvt_f16 <<<wblk, 256, 0, stream>>>(W_in,  w16in);
  k_cvt_f16 <<<wblk, 256, 0, stream>>>(W_out, w16out);
  k_proj_in <<<128,  256, 0, stream>>>(z, w16in, b_in, hid, h16);            // 1024 waves
  k_csq     <<<512,  256, 0, stream>>>(cb, csq, cb16);                       // 4096 waves
  k_argmin  <<<64,   256, 0, stream>>>(h16, cb16, csq, idx);                 // 512 waves
  k_proj_out<<<640,  256, 0, stream>>>(idx, cb16, w16out, b_out, mask, out); // 5120 waves
}
